// PersImage_6828998001468
// MI455X (gfx1250) — matrix operand order repaired and checked by reading, not compiled, not run on hardware
//
#include <hip/hip_runtime.h>
#include <hip/hip_bf16.h>

typedef _Float16 v16h __attribute__((ext_vector_type(16)));
typedef float    v8f  __attribute__((ext_vector_type(8)));

#define DIM       64
#define NPTS      512
#define KCHUNK    128
#define ROWSTRIDE 136   // 128 + 8 halves padding -> 272B row stride, 16B aligned, bank-spread

union FragU {
    v16h     v;
    unsigned u[8];
};

union PackU {
    _Float16 h[2];
    unsigned u;
};

__global__ __launch_bounds__(256) void persimage_wmma(
    const float* __restrict__ h0, const float* __restrict__ h1,
    float* __restrict__ out)
{
    __shared__ _Float16 exT[DIM * ROWSTRIDE];  // A^T staging: exT[d][nn]
    __shared__ _Float16 eyT[DIM * ROWSTRIDE];  // B staging (col-major in e): eyT[e][nn]
    __shared__ float    bp0[NPTS];
    __shared__ float    bp1[NPTS];
    __shared__ int      smax;

    const int tid  = threadIdx.x;
    const int lane = tid & 31;
    const int wave = tid >> 5;
    const int bx   = blockIdx.x;           // 0..63 -> (batch, channel)
    const int b    = bx >> 1;
    const int c    = bx & 1;

    const float* pts = (c == 0 ? h0 : h1) + b * (NPTS * 2);

    if (tid == 0) smax = 0;
    for (int i = tid; i < NPTS; i += 256) {
        float p0 = pts[2 * i];
        float p1 = pts[2 * i + 1];
        bp0[i] = p0;        // birth
        bp1[i] = p1 - p0;   // persistence (death - birth)
    }

    // tile assignment: 4x4 grid of 16x16 tiles; each of 8 waves gets 2 tiles in one row
    const int mt  = wave >> 1;         // 0..3 : d-block
    const int nt0 = (wave & 1) * 2;    // 0 or 2 : e-block
    const int nt1 = nt0 + 1;

    const int laneLo    = lane & 15;
    const int laneHiBit = (lane >> 4) & 1;

    v8f acc0 = {};
    v8f acc1 = {};

    const float inv64 = 1.0f / 64.0f;

    for (int kc = 0; kc < 4; ++kc) {
        __syncthreads();   // previous chunk's WMMA reads complete (also covers bp fill on kc==0)

        // Fill chunk: n in [kc*128, kc*128+128), all 64 coordinates, both matrices.
        // 4096 half2-pairs per matrix, 256 threads -> uniform 16 iterations.
        for (int i = tid; i < DIM * (KCHUNK / 2); i += 256) {
            int d  = i >> 6;            // coordinate index 0..63
            int np = i & 63;            // pair index in chunk
            int n  = kc * KCHUNK + np * 2;
            float cd  = (float)d * inv64;
            float dx0 = bp0[n]     - cd;
            float dx1 = bp0[n + 1] - cd;
            float dy0 = bp1[n]     - cd;
            float dy1 = bp1[n + 1] - cd;
            PackU px, py;
            px.h[0] = (_Float16)__expf(-dx0 * dx0 * 50.0f);  // 1/(2*0.01) = 50
            px.h[1] = (_Float16)__expf(-dx1 * dx1 * 50.0f);
            py.h[0] = (_Float16)__expf(-dy0 * dy0 * 50.0f);
            py.h[1] = (_Float16)__expf(-dy1 * dy1 * 50.0f);
            int nn = np * 2;
            *(unsigned*)&exT[d * ROWSTRIDE + nn] = px.u;
            *(unsigned*)&eyT[d * ROWSTRIDE + nn] = py.u;
        }
        __syncthreads();

        // 4 WMMA K-steps of 32 over this 128-wide chunk
        #pragma unroll
        for (int ks = 0; ks < 4; ++ks) {
            const int kloc = ks * 32;

            // A fragment: 16x32 f16, row M = lane%16 (d), K contiguous: 16*(lane>=16) + element
            FragU a;
            {
                const _Float16* rowA =
                    &exT[(mt * 16 + laneLo) * ROWSTRIDE + kloc + laneHiBit * 16];
                #pragma unroll
                for (int v = 0; v < 4; ++v) {
                    a.u[v]     = *(const unsigned*)(rowA + 2 * v);       // K = {0,2,4,6}+hi16
                    a.u[v + 4] = *(const unsigned*)(rowA + 8 + 2 * v);   // K = {8..14}+hi16
                }
            }

            // B fragments: 32x16 f16, col N = lane%16 (e), K = h + 16*(lane>=16)
            FragU b0f, b1f;
            {
                const _Float16* rowB0 =
                    &eyT[(nt0 * 16 + laneLo) * ROWSTRIDE + kloc + laneHiBit * 16];
                const _Float16* rowB1 =
                    &eyT[(nt1 * 16 + laneLo) * ROWSTRIDE + kloc + laneHiBit * 16];
                #pragma unroll
                for (int v = 0; v < 8; ++v) {
                    b0f.u[v] = *(const unsigned*)(rowB0 + 2 * v);
                    b1f.u[v] = *(const unsigned*)(rowB1 + 2 * v);
                }
            }

            acc0 = __builtin_amdgcn_wmma_f32_16x16x32_f16(
                false, a.v, false, b0f.v, (short)0, acc0, false, false);
            acc1 = __builtin_amdgcn_wmma_f32_16x16x32_f16(
                false, a.v, false, b1f.v, (short)0, acc1, false, false);
        }
    }

    // Per-image max (all accumulators are >= 0: sums of products of Gaussians)
    float m = 0.0f;
    #pragma unroll
    for (int v = 0; v < 8; ++v) {
        m = fmaxf(m, acc0[v]);
        m = fmaxf(m, acc1[v]);
    }
    #pragma unroll
    for (int off = 16; off > 0; off >>= 1)
        m = fmaxf(m, __shfl_xor(m, off, 32));
    if (lane == 0) atomicMax(&smax, __float_as_int(m));   // bit-pattern max valid for >=0 floats
    __syncthreads();

    const float inv = 1.0f / __int_as_float(smax);        // 1/(2*pi*v) scale cancels in im/max(im)

    float* outp = out + (size_t)bx * (DIM * DIM);         // (b,c) image base
    const int e0 = nt0 * 16 + laneLo;
    const int e1 = nt1 * 16 + laneLo;
    #pragma unroll
    for (int v = 0; v < 8; ++v) {
        int d = mt * 16 + v + laneHiBit * 8;              // C/D layout: M = v + 8*(lane>=16)
        outp[d * DIM + e0] = acc0[v] * inv;
        outp[d * DIM + e1] = acc1[v] * inv;
    }
}

extern "C" void kernel_launch(void* const* d_in, const int* in_sizes, int n_in,
                              void* d_out, int out_size, void* d_ws, size_t ws_size,
                              hipStream_t stream) {
    const float* h0  = (const float*)d_in[0];
    const float* h1  = (const float*)d_in[1];
    float*       out = (float*)d_out;
    // 64 workgroups: one per (batch, channel) 64x64 image; 256 threads = 8 wave32.
    persimage_wmma<<<dim3(64), dim3(256), 0, stream>>>(h0, h1, out);
}
